// flow1_65515431133849
// MI455X (gfx1250) — matrix operand order repaired and checked by reading, not compiled, not run on hardware
//
#include <hip/hip_runtime.h>
#include <hip/hip_bf16.h>
#include <math.h>

typedef __attribute__((ext_vector_type(16))) _Float16 v16h;
typedef __attribute__((ext_vector_type(8)))  _Float16 v8h;
typedef __attribute__((ext_vector_type(8)))  float    v8f;

#define Zdim   64
#define Bdim   1024
#define Kdim   128
#define Hdim   100
#define HP     112     // H padded to 7 N-tiles of 16
#define KP     128     // H padded to 4 K-steps of 32 for gemm2/3
#define NFdim  8
#define Xdim   784
#define LOG2PI_F 1.8378770664093453f

// ---------------------------------------------------------------------------
// Fast transcendentals: single v_exp_f32 / v_log_f32 / v_rcp_f32 each.
// fast_tanh uses the overflow-safe form 1 - 2/(e^{2x}+1)  (t=inf -> 1).
// ---------------------------------------------------------------------------
__device__ __forceinline__ float fast_rcp(float x) {
  return __builtin_amdgcn_rcpf(x);
}
__device__ __forceinline__ float fast_tanh(float x) {
  const float t = __expf(2.0f * x);
  return 1.0f - 2.0f * fast_rcp(t + 1.0f);
}

// ---------------------------------------------------------------------------
// WMMA fragment gathers from LDS (16-bit layouts per CDNA5 ISA 7.12.2)
// A 16x32 f16: lanes 0-15 rows M=0..15, halves [0..7]=K(hk..hk+7),
//              [8..15]=K(16+hk..23+hk), hk = 8*(lane>=16)
// B 32x16 f16 (stored [N][K] in LDS): lane n = lane&15,
//              halves [0..7]=K(kk..kk+7), [8..15]=K(16+kk..23+kk), kk = 8*(lane>=16)
// ---------------------------------------------------------------------------
__device__ __forceinline__ v16h load_frag_a(const _Float16* src, int row_base,
                                            int k0, int ld) {
  const int lane = threadIdx.x & 31;
  const int m    = lane & 15;
  const int hk   = (lane >> 4) * 8;
  const _Float16* p = src + (row_base + m) * ld + k0 + hk;
  v8h lo = *(const v8h*)(p);
  v8h hi = *(const v8h*)(p + 16);
  v16h f;
#pragma unroll
  for (int i = 0; i < 8; ++i) { f[i] = lo[i]; f[i + 8] = hi[i]; }
  return f;
}

__device__ __forceinline__ v16h load_frag_b(const _Float16* src, int n_base,
                                            int k0, int ldk) {
  const int lane = threadIdx.x & 31;
  const int n    = lane & 15;
  const int kk   = (lane >> 4) * 8;
  const _Float16* p = src + (n_base + n) * ldk + k0 + kk;
  v8h lo = *(const v8h*)(p);
  v8h hi = *(const v8h*)(p + 16);
  v16h f;
#pragma unroll
  for (int i = 0; i < 8; ++i) { f[i] = lo[i]; f[i + 8] = hi[i]; }
  return f;
}

__device__ __forceinline__ v8f wmma_f16(v16h a, v16h b, v8f c) {
  return __builtin_amdgcn_wmma_f32_16x16x32_f16(false, a, false, b,
                                                (short)0, c, false, false);
}

// ---------------------------------------------------------------------------
// One coupling half-step. HS==0: mask keeps n<32, updates n>=32.
//                         HS==1: mask keeps n>=32, updates n<32.
// z tile lives in zreg[32] in WMMA C layout:
//   zreg[t*8+r] = z[row = wave*16 + 8*(lane>=16) + r][col = (lane&15) + 16*t]
// ---------------------------------------------------------------------------
template <int HS>
__device__ __forceinline__ void half_step(
    float* zreg, float* ldp, _Float16* a16, _Float16* hb,
    const _Float16* w0s, const _Float16* w1s, const _Float16* w2s,
    const float* b0s, const float* b1s, const float* b2s) {
  const int tid   = threadIdx.x;
  const int wave  = tid >> 5;
  const int lane  = tid & 31;
  const int ncol  = lane & 15;
  const int m_off = (lane >> 4) * 8;
  const int lrow  = wave * 16;

  // 1) a16 = f16(z * mask)
#pragma unroll
  for (int t = 0; t < 4; ++t) {
#pragma unroll
    for (int r = 0; r < 8; ++r) {
      const int nc    = t * 16 + ncol;
      const bool keep = (HS == 0) ? (nc < 32) : (nc >= 32);
      a16[(lrow + m_off + r) * Zdim + nc] =
          (_Float16)(keep ? zreg[t * 8 + r] : 0.0f);
    }
  }
  __syncthreads();

  // 2) h = tanh(z1 @ w0^T + b0):  M=16/wave, K=64 (2 steps), N=112 (7 tiles)
  {
    v16h afA0 = load_frag_a(a16, lrow, 0, Zdim);   // hoisted: A reused 7x
    v16h afA1 = load_frag_a(a16, lrow, 32, Zdim);
#pragma unroll
    for (int nt = 0; nt < 7; ++nt) {
      v8f acc = {};
      acc = wmma_f16(afA0, load_frag_b(w0s, nt * 16, 0, Zdim), acc);
      acc = wmma_f16(afA1, load_frag_b(w0s, nt * 16, 32, Zdim), acc);
      const int nc = nt * 16 + ncol;
#pragma unroll
      for (int r = 0; r < 8; ++r) {
        hb[(lrow + m_off + r) * KP + nc] =
            (_Float16)fast_tanh(acc[r] + b0s[nc]);
      }
    }
  }
  __syncthreads();

  // 3) mew = h@w1^T + b1 ; sig = sigmoid(h@w2^T + b2 + 5)
  //    K=128 (4 steps, zero-padded), N=64 (4 tiles)
  {
    v16h afH[4];                                   // hoisted: A reused 4x
#pragma unroll
    for (int ks = 0; ks < 4; ++ks) afH[ks] = load_frag_a(hb, lrow, ks * 32, KP);

#pragma unroll
    for (int nt = 0; nt < 4; ++nt) {
      v8f am = {};
      v8f as = {};
#pragma unroll
      for (int ks = 0; ks < 4; ++ks) {
        am = wmma_f16(afH[ks], load_frag_b(w1s, nt * 16, ks * 32, KP), am);
        as = wmma_f16(afH[ks], load_frag_b(w2s, nt * 16, ks * 32, KP), as);
      }
      const int nc     = nt * 16 + ncol;
      const bool freed = (HS == 0) ? (nc >= 32) : (nc < 32);
#pragma unroll
      for (int r = 0; r < 8; ++r) {
        const float mew = am[r] + b1s[nc];
        const float e   = __expf(-(as[r] + b2s[nc] + 5.0f));
        const float den = 1.0f + e;
        if (freed) {
          zreg[nt * 8 + r] = zreg[nt * 8 + r] * fast_rcp(den) + mew;
          ldp[r] -= __logf(den);  // log(sigmoid) = -log(1+e^{-x})
        }
      }
    }
  }
}

// ---------------------------------------------------------------------------
// Flow kernel: 1024 blocks x 256 threads; each block owns 128 rows of z,
// runs all 8 flow steps (16 half-steps) with z resident in registers.
// Reads z0 / logqz from d_out (written by encoder), writes final z / logp.
// ---------------------------------------------------------------------------
__global__ void __launch_bounds__(256)
flow_kernel(const float* __restrict__ w0g, const float* __restrict__ b0g,
            const float* __restrict__ w1g, const float* __restrict__ b1g,
            const float* __restrict__ w2g, const float* __restrict__ b2g,
            float* __restrict__ out_z, float* __restrict__ out_lp) {
  __shared__ __align__(16) _Float16 a16[128 * Zdim];   // 16 KB  z1 in f16
  __shared__ __align__(16) _Float16 hb[128 * KP];      // 32 KB  h  in f16
  __shared__ __align__(16) _Float16 w0s[HP * Zdim];    // 14 KB  [h][z]
  __shared__ __align__(16) _Float16 w1s[Zdim * KP];    // 16 KB  [z][h-pad]
  __shared__ __align__(16) _Float16 w2s[Zdim * KP];    // 16 KB  [z][h-pad]
  __shared__ float b0s[HP];
  __shared__ float b1s[Zdim];
  __shared__ float b2s[Zdim];

  const int  tid   = threadIdx.x;
  const int  wave  = tid >> 5;
  const int  lane  = tid & 31;
  const int  ncol  = lane & 15;
  const int  m_off = (lane >> 4) * 8;
  const long grow  = (long)blockIdx.x * 128 + wave * 16;

  // zero the permanent K-pad columns of hb (written once, never touched again)
  for (int i = tid; i < 128 * (KP - HP); i += 256) {
    const int r = i / (KP - HP);
    const int c = HP + i % (KP - HP);
    hb[r * KP + c] = (_Float16)0.0f;
  }

  // load z tile into registers (WMMA C layout)
  float zreg[32];
  float ldp[8];
  {
    const float* zp = out_z + (grow + m_off) * Zdim;
#pragma unroll
    for (int t = 0; t < 4; ++t)
#pragma unroll
      for (int r = 0; r < 8; ++r)
        zreg[t * 8 + r] = zp[(long)r * Zdim + t * 16 + ncol];
  }
#pragma unroll
  for (int r = 0; r < 8; ++r) ldp[r] = 0.0f;

  for (int p = 0; p < NFdim; ++p) {
    // stage this step's weights into LDS as f16, zero-padded
    for (int i = tid; i < HP * Zdim; i += 256) {
      const int r = i >> 6, c = i & 63;
      w0s[i] = (_Float16)((r < Hdim) ? w0g[((long)p * Hdim + r) * Zdim + c]
                                     : 0.0f);
    }
    for (int i = tid; i < Zdim * KP; i += 256) {
      const int r = i >> 7, c = i & 127;
      const bool v = (c < Hdim);
      w1s[i] = (_Float16)(v ? w1g[((long)p * Zdim + r) * Hdim + c] : 0.0f);
      w2s[i] = (_Float16)(v ? w2g[((long)p * Zdim + r) * Hdim + c] : 0.0f);
    }
    for (int i = tid; i < HP; i += 256)
      b0s[i] = (i < Hdim) ? b0g[(long)p * Hdim + i] : 0.0f;
    for (int i = tid; i < Zdim; i += 256) {
      b1s[i] = b1g[(long)p * Zdim + i];
      b2s[i] = b2g[(long)p * Zdim + i];
    }
    if (p + 1 < NFdim) {  // gfx1250 global_prefetch path for next step weights
      __builtin_prefetch(w0g + (long)(p + 1) * Hdim * Zdim, 0, 1);
      __builtin_prefetch(w1g + (long)(p + 1) * Zdim * Hdim, 0, 1);
      __builtin_prefetch(w2g + (long)(p + 1) * Zdim * Hdim, 0, 1);
    }
    __syncthreads();

    half_step<0>(zreg, ldp, a16, hb, w0s, w1s, w2s, b0s, b1s, b2s);
    half_step<1>(zreg, ldp, a16, hb, w0s, w1s, w2s, b0s, b1s, b2s);
    __syncthreads();  // protect w*s before next step's reload
  }

  // write back final z
  {
    float* zp = out_z + (grow + m_off) * Zdim;
#pragma unroll
    for (int t = 0; t < 4; ++t)
#pragma unroll
      for (int r = 0; r < 8; ++r)
        zp[(long)r * Zdim + t * 16 + ncol] = zreg[t * 8 + r];
  }

  // reduce logdet across the 16 lanes sharing each row group, write logp
#pragma unroll
  for (int s = 1; s < 16; s <<= 1)
#pragma unroll
    for (int r = 0; r < 8; ++r)
      ldp[r] += __shfl_xor(ldp[r], s, 32);
  if (ncol == 0) {
#pragma unroll
    for (int r = 0; r < 8; ++r) {
      const long kb = grow + m_off + r;
      out_lp[kb] = out_lp[kb] - ldp[r];  // logqz - logdetsum
    }
  }
}

// ---------------------------------------------------------------------------
// Encoder: one block per batch row b. fc1->tanh->fc2->tanh->fc3, then
// z0 = eps*exp(0.5*logvar)+mean for all K, and logqz (= -0.5*(Z*log2pi
// + sum(logvar) + sum(eps^2)) since (z-mean)^2*exp(-logvar) == eps^2).
// ---------------------------------------------------------------------------
__global__ void __launch_bounds__(256)
encoder_kernel(const float* __restrict__ x,    const float* __restrict__ eps,
               const float* __restrict__ fc1w, const float* __restrict__ fc1b,
               const float* __restrict__ fc2w, const float* __restrict__ fc2b,
               const float* __restrict__ fc3w, const float* __restrict__ fc3b,
               float* __restrict__ out_z, float* __restrict__ out_lp) {
  __shared__ float xr[Xdim];
  __shared__ float h1[200];
  __shared__ float h2[200];
  __shared__ float o3[128];
  __shared__ float slv;

  const int b   = blockIdx.x;
  const int tid = threadIdx.x;

  for (int i = tid; i < Xdim; i += 256) xr[i] = x[(long)b * Xdim + i];
  __syncthreads();

  for (int n = tid; n < 200; n += 256) {
    float s = fc1b[n];
    const float* w = fc1w + (long)n * Xdim;
    for (int i = 0; i < Xdim; ++i) s += xr[i] * w[i];
    h1[n] = fast_tanh(s);
  }
  __syncthreads();

  for (int n = tid; n < 200; n += 256) {
    float s = fc2b[n];
    const float* w = fc2w + (long)n * 200;
    for (int i = 0; i < 200; ++i) s += h1[i] * w[i];
    h2[n] = fast_tanh(s);
  }
  __syncthreads();

  for (int n = tid; n < 128; n += 256) {
    float s = fc3b[n];
    const float* w = fc3w + (long)n * 200;
    for (int i = 0; i < 200; ++i) s += h2[i] * w[i];
    o3[n] = s;  // [0:64)=mean, [64:128)=logvar
  }
  __syncthreads();

  if (tid == 0) {
    float s = 0.0f;
    for (int j = 0; j < Zdim; ++j) s += o3[Zdim + j];
    slv = s;
  }
  __syncthreads();

  // z0 for all K
  for (int idx = tid; idx < Kdim * Zdim; idx += 256) {
    const int k = idx >> 6, j = idx & 63;
    const long off = ((long)k * Bdim + b) * Zdim + j;
    const float e  = eps[off];
    out_z[off] = e * __expf(0.5f * o3[Zdim + j]) + o3[j];
  }
  // logqz
  for (int k = tid; k < Kdim; k += 256) {
    float se = 0.0f;
    const float* ep = eps + ((long)k * Bdim + b) * Zdim;
    for (int j = 0; j < Zdim; ++j) se += ep[j] * ep[j];
    out_lp[(long)k * Bdim + b] =
        -0.5f * ((float)Zdim * LOG2PI_F + slv + se);
  }
}

// ---------------------------------------------------------------------------
extern "C" void kernel_launch(void* const* d_in, const int* in_sizes, int n_in,
                              void* d_out, int out_size, void* d_ws,
                              size_t ws_size, hipStream_t stream) {
  (void)in_sizes; (void)n_in; (void)out_size; (void)d_ws; (void)ws_size;
  const float* x    = (const float*)d_in[0];
  const float* eps  = (const float*)d_in[1];
  const float* fc1w = (const float*)d_in[2];
  const float* fc1b = (const float*)d_in[3];
  const float* fc2w = (const float*)d_in[4];
  const float* fc2b = (const float*)d_in[5];
  const float* fc3w = (const float*)d_in[6];
  const float* fc3b = (const float*)d_in[7];
  const float* w0   = (const float*)d_in[8];
  const float* b0   = (const float*)d_in[9];
  const float* w1   = (const float*)d_in[10];
  const float* b1   = (const float*)d_in[11];
  const float* w2   = (const float*)d_in[12];
  const float* b2   = (const float*)d_in[13];

  float* out    = (float*)d_out;
  float* out_z  = out;                                   // (K,B,Z)
  float* out_lp = out + (size_t)Kdim * Bdim * Zdim;      // (K,B)

  encoder_kernel<<<Bdim, 256, 0, stream>>>(x, eps, fc1w, fc1b, fc2w, fc2b,
                                           fc3w, fc3b, out_z, out_lp);
  flow_kernel<<<(Kdim * Bdim) / 128, 256, 0, stream>>>(w0, b0, w1, b1, w2, b2,
                                                       out_z, out_lp);
}
